// GNN_63591285784898
// MI455X (gfx1250) — matrix operand order repaired and checked by reading, not compiled, not run on hardware
//
#include <hip/hip_runtime.h>
#include <hip/hip_bf16.h>

typedef __attribute__((ext_vector_type(2)))  float    v2f;
typedef __attribute__((ext_vector_type(8)))  float    v8f;
typedef __attribute__((ext_vector_type(16))) _Float16 v16h;

#if __has_builtin(__builtin_amdgcn_wmma_f32_16x16x4_f32)
#define USE_F32_WMMA 1
#endif

#define D 512

// ---------------- zero workspace ----------------
__global__ void k_zero(unsigned int* __restrict__ p, size_t n) {
    size_t i = (size_t)blockIdx.x * blockDim.x + threadIdx.x;
    size_t st = (size_t)gridDim.x * blockDim.x;
    for (; i < n; i += st) p[i] = 0u;
}

// ---------------- build G (6x512 edge-gate vectors) into pair-layout Gf ----------------
// Gf[p*32 + n*2 + j] = G[n][2p+j]   (n = 0..15, cols 6..15 zero)
__global__ __launch_bounds__(512) void k_setup_g(
    const float* __restrict__ x2, const float* __restrict__ x3, const float* __restrict__ x4,
    const float* __restrict__ ea1, const float* __restrict__ ea2, const float* __restrict__ ea3,
    const float* __restrict__ We1, const float* __restrict__ be1,
    const float* __restrict__ We2, const float* __restrict__ be2,
    const float* __restrict__ We3, const float* __restrict__ be3,
    const float* __restrict__ We4, const float* __restrict__ be4,
    const float* __restrict__ We5, const float* __restrict__ be5,
    const float* __restrict__ We6, const float* __restrict__ be6,
    float* __restrict__ Gf) {
    const int c = threadIdx.x;              // 0..511
    const float e1 = ea1[0], e2 = ea2[0], e3 = ea3[0];
    float g[6];
    g[0] = x2[0] * fmaxf(e1 * We1[c] + be1[c], 0.f);
    float t = 0.f;
    for (int r = 0; r < 4; ++r) t += x3[r] * fmaxf(e2 * We2[r * D + c] + be2[r * D + c], 0.f);
    g[1] = t * 0.25f;
    t = 0.f;
    for (int r = 0; r < 6; ++r) t += x4[r] * fmaxf(e3 * We3[r * D + c] + be3[r * D + c], 0.f);
    g[2] = t * (1.f / 6.f);
    g[3] = x2[0] * fmaxf(e1 * We4[c] + be4[c], 0.f);
    t = 0.f;
    for (int r = 0; r < 4; ++r) t += x3[r] * fmaxf(e2 * We5[r * D + c] + be5[r * D + c], 0.f);
    g[4] = t * 0.25f;
    t = 0.f;
    for (int r = 0; r < 6; ++r) t += x4[r] * fmaxf(e3 * We6[r * D + c] + be6[r * D + c], 0.f);
    g[5] = t * (1.f / 6.f);
    const int base = (c >> 1) * 32 + (c & 1);
    for (int n = 0; n < 16; ++n) Gf[base + n * 2] = (n < 6) ? g[n] : 0.f;
}

// ---------------- degree histogram over dst ----------------
__global__ void k_hist(const int* __restrict__ dst, int* __restrict__ cnt, int E_) {
    int i = blockIdx.x * blockDim.x + threadIdx.x;
    int st = gridDim.x * blockDim.x;
    for (; i < E_; i += st) atomicAdd(&cnt[dst[i]], 1);
}

// ---------------- W[src] += 1/cnt[dst] ----------------
__global__ void k_weight(const int* __restrict__ src, const int* __restrict__ dst,
                         const int* __restrict__ cnt, float* __restrict__ W, int E_) {
    int i = blockIdx.x * blockDim.x + threadIdx.x;
    int st = gridDim.x * blockDim.x;
    for (; i < E_; i += st) {
        int cv = cnt[dst[i]];
        float w = 1.0f / (float)(cv > 0 ? cv : 1);
        atomicAdd(&W[src[i]], w);
    }
}

// ---------------- v1 = W1^T x1, v3 = W3^T x1, S = {sum W1, sum W3} ----------------
__global__ __launch_bounds__(256) void k_rowsum(
    const float* __restrict__ x1, const float* __restrict__ W1, const float* __restrict__ W3,
    float* __restrict__ v1, float* __restrict__ v3, float* __restrict__ S, int n1) {
    const int cg = threadIdx.x & 127;   // 128 col-groups of 4 floats
    const int rl = threadIdx.x >> 7;    // 2 row lanes per block
    float a1x = 0, a1y = 0, a1z = 0, a1w = 0;
    float a3x = 0, a3y = 0, a3z = 0, a3w = 0;
    float s1 = 0, s3 = 0;
    const int rstride = gridDim.x * 2;
    for (int r = blockIdx.x * 2 + rl; r < n1; r += rstride) {
        const float w1 = W1[r], w3 = W3[r];
        const float4 xv = *reinterpret_cast<const float4*>(x1 + (size_t)r * D + cg * 4);
        a1x += w1 * xv.x; a1y += w1 * xv.y; a1z += w1 * xv.z; a1w += w1 * xv.w;
        a3x += w3 * xv.x; a3y += w3 * xv.y; a3z += w3 * xv.z; a3w += w3 * xv.w;
        if (cg == 0) { s1 += w1; s3 += w3; }
    }
    atomicAdd(&v1[cg * 4 + 0], a1x); atomicAdd(&v1[cg * 4 + 1], a1y);
    atomicAdd(&v1[cg * 4 + 2], a1z); atomicAdd(&v1[cg * 4 + 3], a1w);
    atomicAdd(&v3[cg * 4 + 0], a3x); atomicAdd(&v3[cg * 4 + 1], a3y);
    atomicAdd(&v3[cg * 4 + 2], a3z); atomicAdd(&v3[cg * 4 + 3], a3w);
    if (cg == 0) { atomicAdd(&S[0], s1); atomicAdd(&S[1], s3); }
}

// ---------------- bs = G @ x1^T via WMMA: out[n][row] for n=0..5 ----------------
__global__ __launch_bounds__(256) void k_bs(const float* __restrict__ x1,
                                            const float* __restrict__ Gf,
                                            float* __restrict__ outb, int n1) {
    __shared__ float sG[D * 16];
    for (int i = threadIdx.x; i < D * 16; i += 256) sG[i] = Gf[i];
    __syncthreads();
    const int wave = threadIdx.x >> 5;
    const int lane = threadIdx.x & 31;
    const int m  = lane & 15;   // A row / B,D column index
    const int hi = lane >> 4;
    const int rowBase = (blockIdx.x * 8 + wave) * 16;
    if (rowBase >= n1) return;

    if (rowBase + 16 <= n1) {
        const float* arow = x1 + (size_t)(rowBase + m) * D;
        __builtin_prefetch(arow, 0, 3);                 // global_prefetch_b8
        v8f c = {};
#ifdef USE_F32_WMMA
        for (int k0 = 0; k0 < D; k0 += 4) {
            const int ka = k0 + 2 * hi;                 // A/B frag: VGPR v holds K = v + 2*hi
            v2f a = *reinterpret_cast<const v2f*>(arow + ka);
            v2f b = *reinterpret_cast<const v2f*>(&sG[(ka >> 1) * 32 + m * 2]);
            c = __builtin_amdgcn_wmma_f32_16x16x4_f32(false, a, false, b,
                                                      (short)0, c, false, false);
        }
#else
        for (int k0 = 0; k0 < D; k0 += 32) {
            v16h a, b;
#pragma unroll
            for (int e = 0; e < 8; ++e) {
                a[e]     = (_Float16)arow[k0 + 8 * hi + e];
                a[e + 8] = (_Float16)arow[k0 + 16 + 8 * hi + e];
            }
#pragma unroll
            for (int e = 0; e < 16; ++e) {
                const int k = k0 + 8 * hi + (e & 7) + ((e >> 3) << 4);
                b[e] = (_Float16)sG[(k >> 1) * 32 + m * 2 + (k & 1)];
            }
            c = __builtin_amdgcn_wmma_f32_16x16x32_f16(false, a, false, b,
                                                       (short)0, c, false, false);
        }
#endif
        if (m < 6) {                                    // only the 6 real G columns
            float* col = outb + (size_t)m * n1;
#pragma unroll
            for (int v = 0; v < 8; ++v) col[rowBase + v + 8 * hi] = c[v];
        }
    } else {
        // scalar tail (N1 % 16 != 0 safety; unused for N1 = 100000)
        const int rem = n1 - rowBase;
        if (hi == 0 && m < rem) {
            const float* arow = x1 + (size_t)(rowBase + m) * D;
            float acc[6] = {0, 0, 0, 0, 0, 0};
            for (int k = 0; k < D; ++k) {
                const float av = arow[k];
                const float* gg = &sG[(k >> 1) * 32 + (k & 1)];
#pragma unroll
                for (int n = 0; n < 6; ++n) acc[n] += av * gg[n * 2];
            }
            for (int n = 0; n < 6; ++n) outb[(size_t)n * n1 + rowBase + m] = acc[n];
        }
    }
}

// ---------------- tiny MLP head -> logits ----------------
__global__ __launch_bounds__(64) void k_final(
    const float* __restrict__ x2, const float* __restrict__ x3, const float* __restrict__ x4,
    const float* __restrict__ Wg1, const float* __restrict__ bg1,
    const float* __restrict__ Wg3, const float* __restrict__ bg3,
    const float* __restrict__ Wg4, const float* __restrict__ bg4,
    const float* __restrict__ Wg5, const float* __restrict__ bg5,
    const float* __restrict__ Wg6, const float* __restrict__ bg6,
    const float* __restrict__ Wf1, const float* __restrict__ bf1,
    const float* __restrict__ Wf2, const float* __restrict__ bf2,
    const float* __restrict__ v1, const float* __restrict__ v3,
    const float* __restrict__ S, float* __restrict__ out, int n1) {
    __shared__ float feat[384];
    __shared__ float hb[64];
    const int h = threadIdx.x;  // 64 threads
    const float inv = 1.f / (float)n1;
    float a1m = S[0] * bg1[h];
    for (int c = 0; c < D; ++c) a1m += v1[c] * Wg1[c * 64 + h];
    a1m *= inv;
    float a3m = S[1] * bg3[h];
    for (int c = 0; c < D; ++c) a3m += v3[c] * Wg3[c * 64 + h];
    a3m *= inv;
    const float a4v = x2[0] * Wg4[h] + bg4[h];
    float a5v = bg5[h];
    for (int r = 0; r < 4; ++r) a5v += x3[r] * Wg5[r * 64 + h];
    float a6v = bg6[h];
    for (int r = 0; r < 6; ++r) a6v += x4[r] * Wg6[r * 64 + h];
    feat[h] = a1m; feat[64 + h] = a4v; feat[128 + h] = a3m;
    feat[192 + h] = a4v; feat[256 + h] = a5v; feat[320 + h] = a6v;
    __syncthreads();
    float acc = bf1[h];
    for (int i = 0; i < 384; ++i) acc += feat[i] * Wf1[i * 64 + h];
    hb[h] = fmaxf(acc, 0.f);
    __syncthreads();
    if (h < 2) {
        float l = bf2[h];
        for (int j = 0; j < 64; ++j) l += hb[j] * Wf2[j * 2 + h];
        out[h] = l;
    }
}

// d_in index map (setup_inputs dict order, params flattened in insertion order)
enum {
    I_X1 = 0, I_X2, I_X3, I_X4, I_EI12, I_EI13, I_EI14,
    I_EA1, I_EA2, I_EA3,
    I_WE1, I_BE1, I_WE2, I_BE2, I_WE3, I_BE3, I_WE4, I_BE4, I_WE5, I_BE5, I_WE6, I_BE6,
    I_WG1, I_BG1, I_WG2, I_BG2, I_WG3, I_BG3, I_WG4, I_BG4, I_WG5, I_BG5, I_WG6, I_BG6,
    I_WF1, I_BF1, I_WF2, I_BF2
};

extern "C" void kernel_launch(void* const* d_in, const int* in_sizes, int n_in,
                              void* d_out, int out_size, void* d_ws, size_t ws_size,
                              hipStream_t stream) {
    const float* x1 = (const float*)d_in[I_X1];
    const int n1 = in_sizes[I_X1] / D;
    const int E_ = in_sizes[I_EI12] / 2;
    const int* ei12 = (const int*)d_in[I_EI12];
    const int* ei14 = (const int*)d_in[I_EI14];

    // workspace carve
    char* ws = (char*)d_ws;
    int*   cnt1 = (int*)ws;            ws += (size_t)n1 * 4;
    int*   cnt3 = (int*)ws;            ws += (size_t)n1 * 4;
    float* W1   = (float*)ws;          ws += (size_t)n1 * 4;
    float* W3   = (float*)ws;          ws += (size_t)n1 * 4;
    float* v1   = (float*)ws;          ws += D * 4;
    float* v3   = (float*)ws;          ws += D * 4;
    float* S    = (float*)ws;          ws += 2 * 4;
    float* Gf   = (float*)ws;          ws += D * 16 * 4;

    // zero everything we accumulate into (cnt1..S is contiguous)
    const size_t zwords = (size_t)4 * n1 + 2 * D + 2;
    k_zero<<<256, 256, 0, stream>>>((unsigned int*)d_ws, zwords);

    k_setup_g<<<1, 512, 0, stream>>>(
        (const float*)d_in[I_X2], (const float*)d_in[I_X3], (const float*)d_in[I_X4],
        (const float*)d_in[I_EA1], (const float*)d_in[I_EA2], (const float*)d_in[I_EA3],
        (const float*)d_in[I_WE1], (const float*)d_in[I_BE1],
        (const float*)d_in[I_WE2], (const float*)d_in[I_BE2],
        (const float*)d_in[I_WE3], (const float*)d_in[I_BE3],
        (const float*)d_in[I_WE4], (const float*)d_in[I_BE4],
        (const float*)d_in[I_WE5], (const float*)d_in[I_BE5],
        (const float*)d_in[I_WE6], (const float*)d_in[I_BE6], Gf);

    k_hist<<<1024, 256, 0, stream>>>(ei12 + E_, cnt1, E_);
    k_hist<<<1024, 256, 0, stream>>>(ei14 + E_, cnt3, E_);
    k_weight<<<1024, 256, 0, stream>>>(ei12, ei12 + E_, cnt1, W1, E_);
    k_weight<<<1024, 256, 0, stream>>>(ei14, ei14 + E_, cnt3, W3, E_);

    k_rowsum<<<256, 256, 0, stream>>>(x1, W1, W3, v1, v3, S, n1);

    const int tiles = (n1 + 15) / 16;
    const int blocks = (tiles + 7) / 8;
    k_bs<<<blocks, 256, 0, stream>>>(x1, Gf, (float*)d_out + 2, n1);

    k_final<<<1, 64, 0, stream>>>(
        (const float*)d_in[I_X2], (const float*)d_in[I_X3], (const float*)d_in[I_X4],
        (const float*)d_in[I_WG1], (const float*)d_in[I_BG1],
        (const float*)d_in[I_WG3], (const float*)d_in[I_BG3],
        (const float*)d_in[I_WG4], (const float*)d_in[I_BG4],
        (const float*)d_in[I_WG5], (const float*)d_in[I_BG5],
        (const float*)d_in[I_WG6], (const float*)d_in[I_BG6],
        (const float*)d_in[I_WF1], (const float*)d_in[I_BF1],
        (const float*)d_in[I_WF2], (const float*)d_in[I_BF2],
        v1, v3, S, (float*)d_out, n1);
}
